// Attention_40707700031913
// MI455X (gfx1250) — matrix operand order repaired and checked by reading, not compiled, not run on hardware
//
#include <hip/hip_runtime.h>
#include <math.h>

typedef __attribute__((ext_vector_type(16))) __bf16 v16bf;
typedef __attribute__((ext_vector_type(8)))  float  v8f;
typedef unsigned int u32;

union AFragB { v16bf v; u32 u[8]; };

__device__ __forceinline__ __bf16 f2bf(float f) { return (__bf16)f; }

// Async global->LDS path (ASYNCcnt) if the toolchain exposes it.
#if defined(__gfx1250__) && __has_builtin(__builtin_amdgcn_global_load_async_to_lds_b128)
#define USE_ASYNC 1
typedef __attribute__((__vector_size__(4 * sizeof(int)))) int vi4;
#define ASG __attribute__((address_space(1)))
#define ASL __attribute__((address_space(3)))
template <int N>
__device__ __forceinline__ void wait_async() {
#if __has_builtin(__builtin_amdgcn_s_wait_asynccnt)
  __builtin_amdgcn_s_wait_asynccnt(N);
#else
  asm volatile("s_wait_asynccnt %0" ::"n"(N) : "memory");
#endif
}
__device__ __forceinline__ void async_b128(const __bf16* g, __bf16* l) {
  __builtin_amdgcn_global_load_async_to_lds_b128((ASG vi4*)g, (ASL vi4*)l, 0, 0);
}
__device__ __forceinline__ void async_b128_pair(const __bf16* g, __bf16* l) {
  __builtin_amdgcn_global_load_async_to_lds_b128((ASG vi4*)g, (ASL vi4*)l, 0, 0);
  __builtin_amdgcn_global_load_async_to_lds_b128((ASG vi4*)g, (ASL vi4*)l, 16, 0);
}
#else
#define USE_ASYNC 0
#endif

// LDS strides (bf16 units): 16B-aligned rows (for B128 async) + bank spread
#define AT 40   // A tile, 32 cols + pad   (80 B rows)
#define BT 72   // B tile, 64 cols + pad   (144 B rows)
#define KST 40  // K^T tile, 32 cols + pad
#define VST 72  // V tile, 64 cols + pad
#define PST 34  // P tile (ds path only)

// ---------------------------------------------------------------------------
// LayerNorm (gamma-only, biased variance) -> bf16; one block per 512-wide row
// ---------------------------------------------------------------------------
__global__ __launch_bounds__(256) void ln_to_bf16(const float* __restrict__ x,
                                                  const float* __restrict__ g,
                                                  __bf16* __restrict__ out) {
  const int row = blockIdx.x;
  const float* xr = x + (size_t)row * 512;
  const int t = threadIdx.x;
  float v0 = xr[t], v1 = xr[t + 256];
  float s = v0 + v1, s2 = v0 * v0 + v1 * v1;
  for (int off = 16; off >= 1; off >>= 1) {
    s  += __shfl_xor(s,  off, 32);
    s2 += __shfl_xor(s2, off, 32);
  }
  __shared__ float sh_s[8], sh_s2[8];
  int wv = t >> 5, ln = t & 31;
  if (ln == 0) { sh_s[wv] = s; sh_s2[wv] = s2; }
  __syncthreads();
  if (wv == 0) {
    float a = (ln < 8) ? sh_s[ln] : 0.f;
    float b = (ln < 8) ? sh_s2[ln] : 0.f;
    for (int off = 4; off >= 1; off >>= 1) {
      a += __shfl_xor(a, off, 32);
      b += __shfl_xor(b, off, 32);
    }
    if (ln == 0) { sh_s[0] = a; sh_s2[0] = b; }
  }
  __syncthreads();
  float mean = sh_s[0] * (1.0f / 512.0f);
  float var  = sh_s2[0] * (1.0f / 512.0f) - mean * mean;
  float rstd = rsqrtf(var + 1e-5f);
  out[(size_t)row * 512 + t]       = f2bf((v0 - mean) * rstd * g[t]);
  out[(size_t)row * 512 + t + 256] = f2bf((v1 - mean) * rstd * g[t + 256]);
}

// Final LayerNorm: fp32 in-place (row-exclusive blocks => in-place safe)
__global__ __launch_bounds__(256) void ln_f32_inplace(float* __restrict__ x,
                                                      const float* __restrict__ g) {
  const int row = blockIdx.x;
  float* xr = x + (size_t)row * 512;
  const int t = threadIdx.x;
  float v0 = xr[t], v1 = xr[t + 256];
  float s = v0 + v1, s2 = v0 * v0 + v1 * v1;
  for (int off = 16; off >= 1; off >>= 1) {
    s  += __shfl_xor(s,  off, 32);
    s2 += __shfl_xor(s2, off, 32);
  }
  __shared__ float sh_s[8], sh_s2[8];
  int wv = t >> 5, ln = t & 31;
  if (ln == 0) { sh_s[wv] = s; sh_s2[wv] = s2; }
  __syncthreads();
  if (wv == 0) {
    float a = (ln < 8) ? sh_s[ln] : 0.f;
    float b = (ln < 8) ? sh_s2[ln] : 0.f;
    for (int off = 4; off >= 1; off >>= 1) {
      a += __shfl_xor(a, off, 32);
      b += __shfl_xor(b, off, 32);
    }
    if (ln == 0) { sh_s[0] = a; sh_s2[0] = b; }
  }
  __syncthreads();
  float mean = sh_s[0] * (1.0f / 512.0f);
  float var  = sh_s2[0] * (1.0f / 512.0f) - mean * mean;
  float rstd = rsqrtf(var + 1e-5f);
  xr[t]       = (v0 - mean) * rstd * g[t];
  xr[t + 256] = (v1 - mean) * rstd * g[t + 256];
}

__global__ void cvt_f32_bf16(const float* __restrict__ src, __bf16* __restrict__ dst, int n) {
  int i = blockIdx.x * 256 + threadIdx.x;
  if (i < n) dst[i] = f2bf(src[i]);
}

// ---------------------------------------------------------------------------
// 128x64-tile bf16 WMMA GEMM:  C[M,N] = A[M,K] * B[K,N]
// mode 0: QKV epilogue (scatter to q / k^T / v, q scaled); mode 1: fp32 store
// ---------------------------------------------------------------------------
__global__ __launch_bounds__(256) void gemm_bf16_128x64(
    const __bf16* __restrict__ A, const __bf16* __restrict__ B,
    int M, int N, int K, int mode, float qscale,
    __bf16* __restrict__ qout, __bf16* __restrict__ ktout,
    __bf16* __restrict__ vout, float* __restrict__ fout) {
  __shared__ __bf16 atile[128 * AT];
  __shared__ __bf16 btile[32 * BT];

  const int n0 = blockIdx.x * 64;
  const int m0 = blockIdx.y * 128;
  const int t = threadIdx.x;
  const int w = t >> 5, lane = t & 31;
  const int half = lane >> 4, m = lane & 15;
  const int wr = w >> 1, wc = w & 1;

  v8f c[2][2];
#pragma unroll
  for (int ms = 0; ms < 2; ++ms)
#pragma unroll
    for (int ns = 0; ns < 2; ++ns)
#pragma unroll
      for (int i = 0; i < 8; ++i) c[ms][ns][i] = 0.f;

  for (int k0 = 0; k0 < K; k0 += 32) {
#if USE_ASYNC
    { // A tile: 128 rows x 32 bf16, 32 B/thread via 2x async b128
      int row = t >> 1, seg = t & 1;
      async_b128_pair(A + (size_t)(m0 + row) * K + k0 + seg * 16,
                      atile + row * AT + seg * 16);
    }
    { // B tile: 32 rows x 64 bf16, 16 B/thread via async b128
      int row = t >> 3, seg = t & 7;
      async_b128(B + (size_t)(k0 + row) * N + n0 + seg * 8,
                 btile + row * BT + seg * 8);
    }
    wait_async<0>();
#else
    {
      int row = t >> 1, seg = t & 1;
      const u32* src = (const u32*)(A + (size_t)(m0 + row) * K + k0 + seg * 16);
      u32* dst = (u32*)(atile + row * AT + seg * 16);
#pragma unroll
      for (int j = 0; j < 8; ++j) dst[j] = src[j];
    }
    {
      int row = t >> 3, seg = t & 7;
      const u32* src = (const u32*)(B + (size_t)(k0 + row) * N + n0 + seg * 8);
      u32* dst = (u32*)(btile + row * BT + seg * 8);
#pragma unroll
      for (int j = 0; j < 4; ++j) dst[j] = src[j];
    }
#endif
    __syncthreads();

    AFragB af[2], bf[2];
#pragma unroll
    for (int ms = 0; ms < 2; ++ms) {
      const __bf16* ap = atile + (wr * 32 + ms * 16 + m) * AT;
#pragma unroll
      for (int j = 0; j < 4; ++j) af[ms].u[j]     = *(const u32*)(ap + half * 8 + 2 * j);
#pragma unroll
      for (int j = 0; j < 4; ++j) af[ms].u[4 + j] = *(const u32*)(ap + 16 + half * 8 + 2 * j);
    }
#pragma unroll
    for (int ns = 0; ns < 2; ++ns) {
      const __bf16* bp = btile + wc * 32 + ns * 16 + m;
#pragma unroll
      for (int e = 0; e < 16; ++e) bf[ns].v[e] = bp[((e & 8) * 2 + half * 8 + (e & 7)) * BT];
    }
#pragma unroll
    for (int ms = 0; ms < 2; ++ms)
#pragma unroll
      for (int ns = 0; ns < 2; ++ns)
        c[ms][ns] = __builtin_amdgcn_wmma_f32_16x16x32_bf16(
            false, af[ms].v, false, bf[ns].v, (short)0, c[ms][ns], false, false);
    __syncthreads();
  }

#pragma unroll
  for (int ms = 0; ms < 2; ++ms) {
#pragma unroll
    for (int ns = 0; ns < 2; ++ns) {
#pragma unroll
      for (int r = 0; r < 8; ++r) {
        int grow = m0 + wr * 32 + ms * 16 + r + 8 * half;
        int gcol = n0 + wc * 32 + ns * 16 + m;
        float val = c[ms][ns][r];
        if (mode == 0) {
          int region = gcol >> 9;  // 0=q 1=k 2=v
          int hid = gcol & 511;
          int h = hid >> 6, d = hid & 63;
          int bi = grow >> 11, ni = grow & 2047;
          size_t bh = (size_t)(bi * 8 + h);
          if (region == 0)
            qout[bh * 2048 * 64 + (size_t)ni * 64 + d] = f2bf(val * qscale);
          else if (region == 1)
            ktout[bh * 64 * 2048 + (size_t)d * 2048 + ni] = f2bf(val);
          else
            vout[bh * 2048 * 64 + (size_t)ni * 64 + d] = f2bf(val);
        } else {
          fout[(size_t)grow * N + gcol] = val;
        }
      }
    }
  }
}

// ---------------------------------------------------------------------------
// Flash attention: grid (n/128, b*h); 8 waves * 16 query rows each.
// Double-buffered K/V tiles via async global->LDS (ASYNCcnt) when available.
// ---------------------------------------------------------------------------
#if USE_ASYNC
__device__ __forceinline__ void issue_kv_async(const __bf16* ktbase,
                                               const __bf16* vbase,
                                               __bf16* ktile, __bf16* vtile,
                                               int kv0, int buf, int t) {
  int krow = t >> 2, kseg = t & 3;
  async_b128(ktbase + (size_t)krow * 2048 + kv0 + kseg * 8,
             ktile + buf * (64 * KST) + krow * KST + kseg * 8);
  int vrow = t >> 3, vseg = t & 7;
  async_b128(vbase + (size_t)(kv0 + vrow) * 64 + vseg * 8,
             vtile + buf * (32 * VST) + vrow * VST + vseg * 8);
}
#endif

__global__ __launch_bounds__(256) void flash_attn(
    const __bf16* __restrict__ Q,    // [b*h][2048][64], pre-scaled
    const __bf16* __restrict__ KT,   // [b*h][64][2048]
    const __bf16* __restrict__ V,    // [b*h][2048][64]
    const float* __restrict__ bias,  // [8][2048][2048]
    __bf16* __restrict__ O) {        // [b][2048][512]
  __shared__ __bf16 ktile[2 * 64 * KST];
  __shared__ __bf16 vtile[2 * 32 * VST];
  __shared__ __bf16 ptile[8 * 16 * PST];

  const int bh = blockIdx.y;
  const int bi = bh >> 3, hi = bh & 7;
  const int q0 = blockIdx.x * 128;
  const int t = threadIdx.x, w = t >> 5, lane = t & 31;
  const int half = lane >> 4, m = lane & 15;
  const int qw = q0 + w * 16;

  const __bf16* qbase = Q + (size_t)bh * 2048 * 64;
  const __bf16* ktbase = KT + (size_t)bh * 64 * 2048;
  const __bf16* vbase = V + (size_t)bh * 2048 * 64;
  const float* bbase = bias + (size_t)hi * 2048 * 2048;

  // resident Q fragments: 16 rows x d=64 -> 2 A-frags
  AFragB qa[2];
  {
    const __bf16* qp = qbase + (size_t)(qw + m) * 64;
#pragma unroll
    for (int kf = 0; kf < 2; ++kf) {
#pragma unroll
      for (int j = 0; j < 4; ++j)
        qa[kf].u[j] = *(const u32*)(qp + kf * 32 + half * 8 + 2 * j);
#pragma unroll
      for (int j = 0; j < 4; ++j)
        qa[kf].u[4 + j] = *(const u32*)(qp + kf * 32 + 16 + half * 8 + 2 * j);
    }
  }

  v8f acc[4];
#pragma unroll
  for (int nf = 0; nf < 4; ++nf)
#pragma unroll
    for (int i = 0; i < 8; ++i) acc[nf][i] = 0.f;
  float rm[8], rl[8];
#pragma unroll
  for (int r = 0; r < 8; ++r) { rm[r] = -1e30f; rl[r] = 0.f; }

#if USE_ASYNC
  issue_kv_async(ktbase, vbase, ktile, vtile, 0, 0, t);  // preload tile 0
#endif

  for (int it = 0; it < 64; ++it) {
    const int kv0 = it * 32;
    const int cur = it & 1;
#if USE_ASYNC
    if (it < 63) {
      issue_kv_async(ktbase, vbase, ktile, vtile, kv0 + 32, cur ^ 1, t);
      wait_async<2>();  // tile `it` complete; tile `it+1` in flight
    } else {
      wait_async<0>();
    }
#else
    {
      int row = t >> 2, seg = t & 3;
      const u32* src = (const u32*)(ktbase + (size_t)row * 2048 + kv0 + seg * 8);
      u32* dst = (u32*)(ktile + cur * (64 * KST) + row * KST + seg * 8);
#pragma unroll
      for (int j = 0; j < 4; ++j) dst[j] = src[j];
    }
    {
      int row = t >> 3, seg = t & 7;
      const u32* src = (const u32*)(vbase + (size_t)(kv0 + row) * 64 + seg * 8);
      u32* dst = (u32*)(vtile + cur * (32 * VST) + row * VST + seg * 8);
#pragma unroll
      for (int j = 0; j < 4; ++j) dst[j] = src[j];
    }
#endif
    __syncthreads();  // tile `it` visible to all waves

    const __bf16* kcur = ktile + cur * (64 * KST);
    const __bf16* vcur = vtile + cur * (32 * VST);

    // scores init with pos_bias (loaded directly in C-fragment layout)
    v8f s[2];
#pragma unroll
    for (int nf = 0; nf < 2; ++nf)
#pragma unroll
      for (int r = 0; r < 8; ++r)
        s[nf][r] = bbase[(size_t)(qw + r + 8 * half) * 2048 + kv0 + nf * 16 + m];
    if (kv0 + 32 < 2048)  // prefetch next bias tile -> global_prefetch_b8
      __builtin_prefetch(&bbase[(size_t)(qw + 8 * half) * 2048 + kv0 + 32 + m], 0, 0);

    // S += Q * K^T  (two K-steps of 32 over d=64)
#pragma unroll
    for (int ks = 0; ks < 2; ++ks) {
      AFragB kb[2];
#pragma unroll
      for (int nf = 0; nf < 2; ++nf) {
        const __bf16* bp = kcur + (ks * 32) * KST + nf * 16 + m;
#pragma unroll
        for (int e = 0; e < 16; ++e) kb[nf].v[e] = bp[((e & 8) * 2 + half * 8 + (e & 7)) * KST];
      }
#pragma unroll
      for (int nf = 0; nf < 2; ++nf)
        s[nf] = __builtin_amdgcn_wmma_f32_16x16x32_bf16(
            false, qa[ks].v, false, kb[nf].v, (short)0, s[nf], false, false);
    }

    // online softmax (row = r + 8*half, spread across 16 lanes of one half)
    float pscale[8];
#pragma unroll
    for (int r = 0; r < 8; ++r) {
      float mx = fmaxf(s[0][r], s[1][r]);
#pragma unroll
      for (int off = 8; off >= 1; off >>= 1)
        mx = fmaxf(mx, __shfl_xor(mx, off, 32));
      float mnew = fmaxf(rm[r], mx);
      float sc = __expf(rm[r] - mnew);
      float p0 = __expf(s[0][r] - mnew);
      float p1 = __expf(s[1][r] - mnew);
      float rs = p0 + p1;
#pragma unroll
      for (int off = 8; off >= 1; off >>= 1)
        rs += __shfl_xor(rs, off, 32);
      rm[r] = mnew;
      rl[r] = rl[r] * sc + rs;
      pscale[r] = sc;
      s[0][r] = p0;
      s[1][r] = p1;
    }
#pragma unroll
    for (int nf = 0; nf < 4; ++nf)
#pragma unroll
      for (int r = 0; r < 8; ++r) acc[nf][r] *= pscale[r];

    // C-layout P -> wave-private LDS -> A-layout fragment (bf16)
    __bf16* pw = ptile + w * 16 * PST;
#pragma unroll
    for (int r = 0; r < 8; ++r) {
      pw[(r + 8 * half) * PST + m] = f2bf(s[0][r]);
      pw[(r + 8 * half) * PST + 16 + m] = f2bf(s[1][r]);
    }
    AFragB pa;
#pragma unroll
    for (int j = 0; j < 4; ++j) pa.u[j]     = *(const u32*)(pw + m * PST + half * 8 + 2 * j);
#pragma unroll
    for (int j = 0; j < 4; ++j) pa.u[4 + j] = *(const u32*)(pw + m * PST + 16 + half * 8 + 2 * j);

    // O += P * V
#pragma unroll
    for (int nf = 0; nf < 4; ++nf) {
      AFragB vb;
      const __bf16* vp = vcur + nf * 16 + m;
#pragma unroll
      for (int e = 0; e < 16; ++e) vb.v[e] = vp[((e & 8) * 2 + half * 8 + (e & 7)) * VST];
      acc[nf] = __builtin_amdgcn_wmma_f32_16x16x32_bf16(
          false, pa.v, false, vb.v, (short)0, acc[nf], false, false);
    }
    __syncthreads();  // all waves done reading tile `it` before its buffer is reused
  }

  // normalize and store bf16 to [b][n][h*64+d]
#pragma unroll
  for (int nf = 0; nf < 4; ++nf)
#pragma unroll
    for (int r = 0; r < 8; ++r) {
      int ni = qw + r + 8 * half;
      int d = nf * 16 + m;
      float val = acc[nf][r] / rl[r];
      O[((size_t)bi * 2048 + ni) * 512 + hi * 64 + d] = f2bf(val);
    }
}

// ---------------------------------------------------------------------------
extern "C" void kernel_launch(void* const* d_in, const int* in_sizes, int n_in,
                              void* d_out, int out_size, void* d_ws, size_t ws_size,
                              hipStream_t stream) {
  (void)in_sizes; (void)n_in; (void)out_size; (void)ws_size;
  const float* x        = (const float*)d_in[0];
  const float* pos_bias = (const float*)d_in[1];
  const float* w_qkv    = (const float*)d_in[2];
  const float* w_out    = (const float*)d_in[3];
  const float* g_norm   = (const float*)d_in[4];
  const float* g_out    = (const float*)d_in[5];
  float* out = (float*)d_out;

  char* ws = (char*)d_ws;
  __bf16* xn    = (__bf16*)(ws);                 // 8192*512*2       = 8 MB
  __bf16* wqkv  = (__bf16*)(ws + 8388608);       // 512*1536*2       = 1.5 MB
  __bf16* wout  = (__bf16*)(ws + 9961472);       // 512*512*2        = 0.5 MB
  __bf16* q     = (__bf16*)(ws + 10485760);      // 32*2048*64*2     = 8 MB
  __bf16* kt    = (__bf16*)(ws + 18874368);      // 32*64*2048*2     = 8 MB
  __bf16* v     = (__bf16*)(ws + 27262976);      // 32*2048*64*2     = 8 MB
  __bf16* attno = (__bf16*)(ws + 35651584);      // 8192*512*2       = 8 MB

  ln_to_bf16<<<8192, 256, 0, stream>>>(x, g_norm, xn);
  cvt_f32_bf16<<<(512 * 1536 + 255) / 256, 256, 0, stream>>>(w_qkv, wqkv, 512 * 1536);
  cvt_f32_bf16<<<(512 * 512 + 255) / 256, 256, 0, stream>>>(w_out, wout, 512 * 512);

  // QKV projection: [8192,512] x [512,1536]
  gemm_bf16_128x64<<<dim3(1536 / 64, 8192 / 128), 256, 0, stream>>>(
      xn, wqkv, 8192, 1536, 512, /*mode=*/0, /*qscale=*/0.125f, q, kt, v, nullptr);

  // flash attention
  flash_attn<<<dim3(2048 / 128, 32), 256, 0, stream>>>(q, kt, v, pos_bias, attno);

  // out projection: [8192,512] x [512,512] -> fp32 into d_out
  gemm_bf16_128x64<<<dim3(512 / 64, 8192 / 128), 256, 0, stream>>>(
      attno, wout, 8192, 512, 512, /*mode=*/1, 1.0f, nullptr, nullptr, nullptr, out);

  // final LayerNorm in place on d_out
  ln_f32_inplace<<<8192, 256, 0, stream>>>(out, g_out);
}
